// EdgeSheafLaplacianNetwork_81698867905236
// MI455X (gfx1250) — compile-verified, matrix operand order checked by reading, not run on hardware
//
#include <hip/hip_runtime.h>
#include <hip/hip_bf16.h>
#include <math.h>

typedef __attribute__((ext_vector_type(16))) _Float16 v16h;
typedef __attribute__((ext_vector_type(8)))  float    v8f;

#define N_NODES   2048
#define PAIRS     32768          // N_NODES * 2 * K_NBR
#define NBR       16             // pairs per node, contiguous [16i,16i+16)
#define FEAT      64
#define ROWS      (N_NODES * 4)  // 8192 (n * d)

// ---------------------------------------------------------------------------
// WMMA fragment-packed LDS layout (CDNA5 ISA 16-bit 16x32 layout):
// A-tile (16x32): lane = m + 16*((kk>>3)&1), elem = 2*(4*(kk>>4)+((kk&7)>>1)) + (kk&1)
// B-tile (32x16): identical with column n in place of row m.
// A tile is stored as [lane][16 halves] => fragment read = one contiguous v16h
// (2 x ds_load_b128).  Writers scatter with ds_store_b16 once per tile.
// ---------------------------------------------------------------------------
static __device__ __forceinline__ int frag_pos(int rc, int kk) {
  const int lane = rc + 16 * ((kk >> 3) & 1);
  const int elem = 2 * (4 * (kk >> 4) + ((kk & 7) >> 1)) + (kk & 1);
  return lane * 16 + elem;
}

static __device__ __forceinline__ v16h read_frag(const _Float16* p, int lane) {
  return *(const v16h*)(p + lane * 16);  // 32B aligned, contiguous per lane
}

// Build an A fragment for a 16x32 tile directly from a row-major f32 row:
// lane needs cols [8hi, 8hi+8) then [16+8hi, 16+8hi+8), in element order.
static __device__ __forceinline__ v16h a_frag_global(const float* row, int hi) {
  const float4* r1 = (const float4*)(row + 8 * hi);
  const float4* r2 = (const float4*)(row + 16 + 8 * hi);
  const float4 x0 = r1[0], x1 = r1[1], y0 = r2[0], y1 = r2[1];
  v16h a;
  a[0]  = (_Float16)x0.x; a[1]  = (_Float16)x0.y;
  a[2]  = (_Float16)x0.z; a[3]  = (_Float16)x0.w;
  a[4]  = (_Float16)x1.x; a[5]  = (_Float16)x1.y;
  a[6]  = (_Float16)x1.z; a[7]  = (_Float16)x1.w;
  a[8]  = (_Float16)y0.x; a[9]  = (_Float16)y0.y;
  a[10] = (_Float16)y0.z; a[11] = (_Float16)y0.w;
  a[12] = (_Float16)y1.x; a[13] = (_Float16)y1.y;
  a[14] = (_Float16)y1.z; a[15] = (_Float16)y1.w;
  return a;
}

static __device__ __forceinline__ v8f wmma_f16(v16h a, v16h b, v8f c) {
  return __builtin_amdgcn_wmma_f32_16x16x32_f16(false, a, false, b,
                                                (short)0, c, false, false);
}

// ---------------------------------------------------------------------------
// Kernel 1: edge MLP -> R (P x 4 x 4).  One wave = one 16-pair tile.
//   h0 = [ef[i], ef[j]] (16x128) -> relu(h0@W1+b1) (16x64)
//   -> relu(@W2+b2) (16x32) -> @W3+b3 (16x16) = R tile.  21 v_wmma / wave.
// Layer-1 A fragments come straight from global (float4 runs, no LDS).
// ---------------------------------------------------------------------------
__global__ void __launch_bounds__(128)
mlp_R_kernel(const float* __restrict__ ef, const int* __restrict__ eidx,
             const float* __restrict__ W1, const float* __restrict__ b1,
             const float* __restrict__ W2, const float* __restrict__ b2,
             const float* __restrict__ W3, const float* __restrict__ b3,
             float* __restrict__ R) {
  __shared__ __align__(32) _Float16 sW1p[16 * 512];  // (kt,nt) 4x4 tiles
  __shared__ __align__(32) _Float16 sW2p[4 * 512];   // (kt,nt) 2x2 tiles
  __shared__ __align__(32) _Float16 sW3p[512];       // 1 tile
  __shared__ __align__(32) _Float16 sH1p[4][2 * 512];// per wave, kt 0..1
  __shared__ __align__(32) _Float16 sH2p[4][512];    // per wave

  const int tid  = threadIdx.x;
  const int lane = tid & 31;
  const int wv   = tid >> 5;

  // pack weights into fragment layout
  for (int idx = tid; idx < 128 * 64; idx += 128) {
    const int k = idx >> 6, n = idx & 63;
    sW1p[((k >> 5) * 4 + (n >> 4)) * 512 + frag_pos(n & 15, k & 31)] =
        (_Float16)W1[idx];
  }
  for (int idx = tid; idx < 64 * 32; idx += 128) {
    const int k = idx >> 5, n = idx & 31;
    sW2p[((k >> 5) * 2 + (n >> 4)) * 512 + frag_pos(n & 15, k & 31)] =
        (_Float16)W2[idx];
  }
  for (int idx = tid; idx < 32 * 16; idx += 128) {
    const int k = idx >> 4, n = idx & 15;
    sW3p[frag_pos(n, k)] = (_Float16)W3[idx];
  }
  __syncthreads();

  const int pbase = (blockIdx.x * 4 + wv) * 16;
  const int nl = lane & 15;
  const int mg = lane >> 4;
  const v8f zf = {};

  // layer-1 A fragments direct from global: row m = lane&15
  const int p  = pbase + nl;  // (nl == lane&15 == fragment row m)
  const int ri = eidx[2 * p];
  const int rj = eidx[2 * p + 1];
  const float* rowi = ef + ri * FEAT;
  const float* rowj = ef + rj * FEAT;
  v16h afr[4];
  afr[0] = a_frag_global(rowi,      mg);
  afr[1] = a_frag_global(rowi + 32, mg);
  afr[2] = a_frag_global(rowj,      mg);
  afr[3] = a_frag_global(rowj + 32, mg);

  // ---- layer 1: (16x128)@(128x64) ----
  v8f acc1[4] = {zf, zf, zf, zf};
#pragma unroll
  for (int kt = 0; kt < 4; ++kt) {
#pragma unroll
    for (int nt = 0; nt < 4; ++nt) {
      v16h b = read_frag(&sW1p[(kt * 4 + nt) * 512], lane);
      acc1[nt] = wmma_f16(afr[kt], b, acc1[nt]);
    }
  }
#pragma unroll
  for (int nt = 0; nt < 4; ++nt) {
    const float bias = b1[nt * 16 + nl];
#pragma unroll
    for (int r = 0; r < 8; ++r) {
      const int m = r + 8 * mg;
      const int n = nt * 16 + nl;
      const float v = fmaxf(acc1[nt][r] + bias, 0.0f);
      sH1p[wv][(n >> 5) * 512 + frag_pos(m, n & 31)] = (_Float16)v;
    }
  }
  __syncthreads();

  // ---- layer 2: (16x64)@(64x32) ----
  v8f acc2[2] = {zf, zf};
#pragma unroll
  for (int kt = 0; kt < 2; ++kt) {
    v16h a = read_frag(&sH1p[wv][kt * 512], lane);
#pragma unroll
    for (int nt = 0; nt < 2; ++nt) {
      v16h b = read_frag(&sW2p[(kt * 2 + nt) * 512], lane);
      acc2[nt] = wmma_f16(a, b, acc2[nt]);
    }
  }
#pragma unroll
  for (int nt = 0; nt < 2; ++nt) {
    const float bias = b2[nt * 16 + nl];
#pragma unroll
    for (int r = 0; r < 8; ++r) {
      const int m = r + 8 * mg;
      const int n = nt * 16 + nl;
      const float v = fmaxf(acc2[nt][r] + bias, 0.0f);
      sH2p[wv][frag_pos(m, n)] = (_Float16)v;
    }
  }
  __syncthreads();

  // ---- layer 3: (16x32)@(32x16) -> R tile ----
  v8f acc3 = zf;
  {
    v16h a = read_frag(sH2p[wv], lane);
    v16h b = read_frag(sW3p, lane);
    acc3 = wmma_f16(a, b, acc3);
  }
  const float bias3 = b3[nl];
#pragma unroll
  for (int r = 0; r < 8; ++r) {
    const int m = r + 8 * mg;
    R[(pbase + m) * 16 + nl] = acc3[r] + bias3;  // R[p][k][a] flat = p*16+k*4+a
  }
}

// ---------------------------------------------------------------------------
// Kernel 2: diag[i] = sum over this node's 16 contiguous pairs of R^T R
// ---------------------------------------------------------------------------
__global__ void diag_kernel(const float* __restrict__ R, float* __restrict__ diag) {
  const int i = blockIdx.x * blockDim.x + threadIdx.x;
  if (i >= N_NODES) return;
  float acc[16];
#pragma unroll
  for (int e = 0; e < 16; ++e) acc[e] = 0.0f;
  for (int t = 0; t < NBR; ++t) {
    const float4* r4 = (const float4*)(R + (i * NBR + t) * 16);
    float rk[16];
#pragma unroll
    for (int q = 0; q < 4; ++q) {
      float4 v = r4[q];
      rk[q * 4 + 0] = v.x; rk[q * 4 + 1] = v.y;
      rk[q * 4 + 2] = v.z; rk[q * 4 + 3] = v.w;
    }
#pragma unroll
    for (int k = 0; k < 4; ++k)
#pragma unroll
      for (int a = 0; a < 4; ++a)
#pragma unroll
        for (int b = 0; b < 4; ++b)
          acc[a * 4 + b] += rk[k * 4 + a] * rk[k * 4 + b];
  }
#pragma unroll
  for (int e = 0; e < 16; ++e) diag[i * 16 + e] = acc[e];
}

// ---------------------------------------------------------------------------
// Kernel 3: Dinv = ((diag+diag^T)/2 + eps I)^(-1/2) via 4x4 cyclic Jacobi
// ---------------------------------------------------------------------------
__global__ void dinv_kernel(const float* __restrict__ diag, float* __restrict__ Dinv) {
  const int i = blockIdx.x * blockDim.x + threadIdx.x;
  if (i >= N_NODES) return;
  const float eps = 1.0e-4f;
  float A[4][4], V[4][4];
#pragma unroll
  for (int r = 0; r < 4; ++r)
#pragma unroll
    for (int c = 0; c < 4; ++c) {
      A[r][c] = 0.5f * (diag[i * 16 + r * 4 + c] + diag[i * 16 + c * 4 + r]);
      V[r][c] = (r == c) ? 1.0f : 0.0f;
    }
#pragma unroll
  for (int r = 0; r < 4; ++r) A[r][r] += eps;

  for (int sweep = 0; sweep < 8; ++sweep) {
#pragma unroll
    for (int p = 0; p < 3; ++p) {
#pragma unroll
      for (int q = p + 1; q < 4; ++q) {
        const float apq = A[p][q];
        if (fabsf(apq) < 1.0e-20f) continue;
        const float theta = 0.5f * (A[q][q] - A[p][p]) / apq;
        const float t = (theta >= 0.0f ? 1.0f : -1.0f) /
                        (fabsf(theta) + sqrtf(theta * theta + 1.0f));
        const float c = rsqrtf(t * t + 1.0f);
        const float s = t * c;
#pragma unroll
        for (int k = 0; k < 4; ++k) {
          const float akp = A[k][p], akq = A[k][q];
          A[k][p] = c * akp - s * akq;
          A[k][q] = s * akp + c * akq;
        }
#pragma unroll
        for (int k = 0; k < 4; ++k) {
          const float apk = A[p][k], aqk = A[q][k];
          A[p][k] = c * apk - s * aqk;
          A[q][k] = s * apk + c * aqk;
        }
#pragma unroll
        for (int k = 0; k < 4; ++k) {
          const float vkp = V[k][p], vkq = V[k][q];
          V[k][p] = c * vkp - s * vkq;
          V[k][q] = s * vkp + c * vkq;
        }
      }
    }
  }
  float wi[4];
#pragma unroll
  for (int k = 0; k < 4; ++k) wi[k] = rsqrtf(fmaxf(A[k][k], eps));
#pragma unroll
  for (int r = 0; r < 4; ++r)
#pragma unroll
    for (int c = 0; c < 4; ++c) {
      float v = 0.0f;
#pragma unroll
      for (int k = 0; k < 4; ++k) v += V[r][k] * wi[k] * V[c][k];
      Dinv[i * 16 + r * 4 + c] = v;
    }
}

// ---------------------------------------------------------------------------
// Kernel 4a: X_collapsed[i][f] = mean_d X[i*4+d][f]
// ---------------------------------------------------------------------------
__global__ void xc_kernel(const float* __restrict__ X, float* __restrict__ Xc) {
  const int idx = blockIdx.x * blockDim.x + threadIdx.x;
  if (idx >= N_NODES * FEAT) return;
  const int f = idx & 63, i = idx >> 6;
  const float* x = X + (i * 4) * FEAT + f;
  Xc[idx] = 0.25f * (x[0] + x[FEAT] + x[2 * FEAT] + x[3 * FEAT]);
}

// Kernel 4b: rowsum[i] = sum_j |L1[i,j]| over the 16 known nonzeros
__global__ void rowsum_kernel(const float* __restrict__ L1,
                              const int* __restrict__ eidx,
                              float* __restrict__ rowsum) {
  const int i = blockIdx.x * blockDim.x + threadIdx.x;
  if (i >= N_NODES) return;
  float s = 0.0f;
  for (int t = 0; t < NBR; ++t) {
    const int j = eidx[2 * (i * NBR + t) + 1];
    s += fabsf(L1[i * N_NODES + j]);
  }
  rowsum[i] = s;
}

// ---------------------------------------------------------------------------
// Kernel 5: shared WMMA GEMM  out[M x 64] = stage(A)[M x 64] @ W[64 x 64] (+bias)
//   mode 0: A = (W_sheaf_1 applied per-stalk to X)        -> X_t
//   mode 1: A = L1_norm @ Xc + Xc (sparse 16-nbr gather)  -> hodge
//   mode 2: A = X                                         -> X_res (to out)
// 8 waves/block, 16 rows/wave, 8 v_wmma per wave; fragment-packed LDS.
// ---------------------------------------------------------------------------
__global__ void __launch_bounds__(256)
gemm64_kernel(int mode, int M,
              const float* __restrict__ Asrc, const float* __restrict__ W,
              const float* __restrict__ bias, const float* __restrict__ Ws1,
              const float* __restrict__ L1,   const float* __restrict__ rowsum,
              const int* __restrict__ eidx,   float* __restrict__ out) {
  __shared__ __align__(32) _Float16 sBp[8 * 512];       // (kt 2) x (nt 4) tiles
  __shared__ __align__(32) _Float16 sAp[8][2 * 512];    // per wave, kt 0..1

  const int tid  = threadIdx.x;
  const int lane = tid & 31;
  const int wv   = tid >> 5;

  for (int idx = tid; idx < 64 * 64; idx += 256) {
    const int k = idx >> 6, n = idx & 63;
    sBp[((k >> 5) * 4 + (n >> 4)) * 512 + frag_pos(n & 15, k & 31)] =
        (_Float16)W[idx];
  }

  const int rbase = blockIdx.x * 128 + wv * 16;
  {
    const int m    = lane >> 1;
    const int half = lane & 1;            // = kt of the 32 columns staged
    const int g    = rbase + m;
    float v[32];
    if (g < M) {
      if (mode == 0) {
        const int node = g >> 2, a = g & 3;
        const float w0 = Ws1[a * 4 + 0], w1 = Ws1[a * 4 + 1];
        const float w2 = Ws1[a * 4 + 2], w3 = Ws1[a * 4 + 3];
        const float* x0 = Asrc + (node * 4 + 0) * FEAT + half * 32;
        const float* x1 = Asrc + (node * 4 + 1) * FEAT + half * 32;
        const float* x2 = Asrc + (node * 4 + 2) * FEAT + half * 32;
        const float* x3 = Asrc + (node * 4 + 3) * FEAT + half * 32;
#pragma unroll
        for (int c = 0; c < 32; ++c)
          v[c] = w0 * x0[c] + w1 * x1[c] + w2 * x2[c] + w3 * x3[c];
      } else if (mode == 1) {
        const int i = g;
        const float inv = 1.0f / (rowsum[i] + 1.0e-8f);
        const float* xi = Asrc + i * FEAT + half * 32;
#pragma unroll
        for (int c = 0; c < 32; ++c) v[c] = xi[c];
        for (int t = 0; t < NBR; ++t) {
          const int j = eidx[2 * (i * NBR + t) + 1];
          const float wgt = fabsf(L1[i * N_NODES + j]) * inv;
          const float* xj = Asrc + j * FEAT + half * 32;
#pragma unroll
          for (int c = 0; c < 32; ++c) v[c] += wgt * xj[c];
        }
      } else {
        const float* x = Asrc + g * FEAT + half * 32;
#pragma unroll
        for (int c = 0; c < 32; ++c) v[c] = x[c];
      }
      _Float16* dst = &sAp[wv][half * 512];
#pragma unroll
      for (int c = 0; c < 32; ++c) dst[frag_pos(m, c)] = (_Float16)v[c];
    }
  }
  __syncthreads();

  const v8f zf = {};
  v8f acc[4] = {zf, zf, zf, zf};
#pragma unroll
  for (int kt = 0; kt < 2; ++kt) {
    v16h a = read_frag(&sAp[wv][kt * 512], lane);
#pragma unroll
    for (int nt = 0; nt < 4; ++nt) {
      v16h b = read_frag(&sBp[(kt * 4 + nt) * 512], lane);
      acc[nt] = wmma_f16(a, b, acc[nt]);
    }
  }
  const int nl = lane & 15, mg = lane >> 4;
#pragma unroll
  for (int nt = 0; nt < 4; ++nt) {
    const int n = nt * 16 + nl;
    const float bb = bias ? bias[n] : 0.0f;
#pragma unroll
    for (int r = 0; r < 8; ++r) {
      const int g = rbase + r + 8 * mg;
      if (g < M) out[g * FEAT + n] = acc[nt][r] + bb;
    }
  }
}

// ---------------------------------------------------------------------------
// Kernel 6: Z[(i,a),f] = sum_b Dinv[i][a][b] * X_t[(i,b),f]
// ---------------------------------------------------------------------------
__global__ void z_kernel(const float* __restrict__ Dinv, const float* __restrict__ Xt,
                         float* __restrict__ Z) {
  const int idx = blockIdx.x * blockDim.x + threadIdx.x;
  if (idx >= ROWS * FEAT) return;
  const int f = idx & 63, g = idx >> 6, node = g >> 2, a = g & 3;
  const float* di = Dinv + node * 16 + a * 4;
  float z = 0.0f;
#pragma unroll
  for (int b = 0; b < 4; ++b) z += di[b] * Xt[((node << 2) + b) * FEAT + f];
  Z[idx] = z;
}

// ---------------------------------------------------------------------------
// Kernel 7: block-sparse sheaf.
//   sheaf_i = Dinv_i @ ( diag_i @ Z_i + sum_t off_t @ Z_{j_t} )
//   off_t = -sign(L1[i,j]) * R[ji]^T R[p]
// One block (64 threads = 64 feature columns) per node.
// ---------------------------------------------------------------------------
__global__ void __launch_bounds__(64)
sheaf_kernel(const float* __restrict__ R, const float* __restrict__ diag,
             const float* __restrict__ Dinv, const float* __restrict__ Z,
             const float* __restrict__ L1, const int* __restrict__ eidx,
             const int* __restrict__ ji_idx, float* __restrict__ sheaf) {
  __shared__ float sOff[NBR * 16];
  __shared__ float sDiag[16];
  __shared__ float sDinv[16];
  __shared__ int   sJ[NBR];

  const int i   = blockIdx.x;
  const int tid = threadIdx.x;

  if (tid < NBR) {
    const int p  = i * NBR + tid;
    const int j  = eidx[2 * p + 1];
    const int jp = ji_idx[p];
    sJ[tid] = j;
    const float l = L1[i * N_NODES + j];
    const float sgn = (l > 0.0f ? 1.0f : 0.0f) - (l < 0.0f ? 1.0f : 0.0f);
    const float* Rp  = R + p * 16;
    const float* Rjp = R + jp * 16;
#pragma unroll
    for (int a = 0; a < 4; ++a)
#pragma unroll
      for (int b = 0; b < 4; ++b) {
        float v = 0.0f;
#pragma unroll
        for (int k = 0; k < 4; ++k) v += Rjp[k * 4 + a] * Rp[k * 4 + b];
        sOff[tid * 16 + a * 4 + b] = -sgn * v;
      }
  } else if (tid < 32) {
    sDiag[tid - 16] = diag[i * 16 + (tid - 16)];
  } else if (tid < 48) {
    sDinv[tid - 32] = Dinv[i * 16 + (tid - 32)];
  }
  __syncthreads();

  const int f = tid;
  float zi[4], acc[4];
#pragma unroll
  for (int b = 0; b < 4; ++b) zi[b] = Z[((i << 2) + b) * FEAT + f];
#pragma unroll
  for (int a = 0; a < 4; ++a) {
    float v = 0.0f;
#pragma unroll
    for (int b = 0; b < 4; ++b) v += sDiag[a * 4 + b] * zi[b];
    acc[a] = v;
  }
  for (int t = 0; t < NBR; ++t) {
    const int j = sJ[t];
    float zj[4];
#pragma unroll
    for (int b = 0; b < 4; ++b) zj[b] = Z[((j << 2) + b) * FEAT + f];
#pragma unroll
    for (int a = 0; a < 4; ++a) {
      float v = acc[a];
#pragma unroll
      for (int b = 0; b < 4; ++b) v += sOff[t * 16 + a * 4 + b] * zj[b];
      acc[a] = v;
    }
  }
#pragma unroll
  for (int a = 0; a < 4; ++a) {
    float v = 0.0f;
#pragma unroll
    for (int b = 0; b < 4; ++b) v += sDinv[a * 4 + b] * acc[b];
    sheaf[((i << 2) + a) * FEAT + f] = v;
  }
}

// ---------------------------------------------------------------------------
// Kernel 8: out = X_res - sigmoid(alpha)*repeat(hodge,4) - sigmoid(beta)*elu(sheaf)
// ---------------------------------------------------------------------------
__global__ void final_kernel(const float* __restrict__ hodge,
                             const float* __restrict__ sheaf,
                             const float* __restrict__ alpha,
                             const float* __restrict__ beta,
                             float* __restrict__ out) {
  const int idx = blockIdx.x * blockDim.x + threadIdx.x;
  if (idx >= ROWS * FEAT) return;
  const float sa = 1.0f / (1.0f + expf(-alpha[0]));
  const float sb = 1.0f / (1.0f + expf(-beta[0]));
  const int f = idx & 63, g = idx >> 6;
  const float s = sheaf[idx];
  const float e = (s > 0.0f) ? s : (expf(s) - 1.0f);
  out[idx] = out[idx] - sa * hodge[(g >> 2) * FEAT + f] - sb * e;
}

// ---------------------------------------------------------------------------
extern "C" void kernel_launch(void* const* d_in, const int* in_sizes, int n_in,
                              void* d_out, int out_size, void* d_ws, size_t ws_size,
                              hipStream_t stream) {
  (void)in_sizes; (void)n_in; (void)out_size; (void)ws_size;
  const float* ef     = (const float*)d_in[0];
  const float* L1     = (const float*)d_in[1];
  const float* X      = (const float*)d_in[2];
  const float* W1     = (const float*)d_in[3];
  const float* b1     = (const float*)d_in[4];
  const float* W2     = (const float*)d_in[5];
  const float* b2     = (const float*)d_in[6];
  const float* W3     = (const float*)d_in[7];
  const float* b3     = (const float*)d_in[8];
  const float* Whodge = (const float*)d_in[9];
  const float* Ws1    = (const float*)d_in[10];
  const float* Ws2    = (const float*)d_in[11];
  const float* resW   = (const float*)d_in[12];
  const float* resb   = (const float*)d_in[13];
  const float* alpha  = (const float*)d_in[14];
  const float* beta   = (const float*)d_in[15];
  const int*   eidx   = (const int*)d_in[16];
  const int*   jiidx  = (const int*)d_in[17];
  float* out = (float*)d_out;

  // workspace layout (floats)
  float* ws     = (float*)d_ws;
  float* R      = ws;                      // PAIRS*16      = 524288
  float* diag   = R      + PAIRS * 16;     // N_NODES*16    =  32768
  float* Dinv   = diag   + N_NODES * 16;   //               =  32768
  float* Xc     = Dinv   + N_NODES * 16;   // N_NODES*64    = 131072
  float* rowsum = Xc     + N_NODES * FEAT; //               =   2048
  float* Xt     = rowsum + N_NODES;        // ROWS*64       = 524288
  float* Z      = Xt     + ROWS * FEAT;    //               = 524288
  float* hodge  = Z      + ROWS * FEAT;    // N_NODES*64    = 131072
  float* sheaf  = Xt;                      // alias: X_t dead once Z is built

  // 1) edge MLP -> R (WMMA)
  mlp_R_kernel<<<PAIRS / 64, 128, 0, stream>>>(ef, eidx, W1, b1, W2, b2, W3, b3, R);
  // 2) diag = segment_sum(R^T R)
  diag_kernel<<<N_NODES / 256, 256, 0, stream>>>(R, diag);
  // 3) Dinv = Db^{-1/2}
  dinv_kernel<<<N_NODES / 256, 256, 0, stream>>>(diag, Dinv);
  // 4) X_collapsed, sparse rowsum
  xc_kernel<<<(N_NODES * FEAT) / 256, 256, 0, stream>>>(X, Xc);
  rowsum_kernel<<<N_NODES / 256, 256, 0, stream>>>(L1, eidx, rowsum);
  // 5) X_t = (Ws1 . X) @ Ws2          (WMMA, mode 0)
  gemm64_kernel<<<ROWS / 128, 256, 0, stream>>>(0, ROWS, X, Ws2, nullptr, Ws1,
                                                nullptr, nullptr, nullptr, Xt);
  // 6) hodge = (L1n@Xc + Xc) @ W_hodge (WMMA, mode 1)
  gemm64_kernel<<<N_NODES / 128, 256, 0, stream>>>(1, N_NODES, Xc, Whodge, nullptr,
                                                   nullptr, L1, rowsum, eidx, hodge);
  // 7) X_res = X @ res_W + res_b -> out (WMMA, mode 2)
  gemm64_kernel<<<ROWS / 128, 256, 0, stream>>>(2, ROWS, X, resW, resb, nullptr,
                                                nullptr, nullptr, nullptr, out);
  // 8) Z = Dinv . X_t
  z_kernel<<<(ROWS * FEAT) / 256, 256, 0, stream>>>(Dinv, Xt, Z);
  // 9) block-sparse sheaf = Delta_F @ X_t
  sheaf_kernel<<<N_NODES, 64, 0, stream>>>(R, diag, Dinv, Z, L1, eidx, jiidx, sheaf);
  // 10) combine
  final_kernel<<<(ROWS * FEAT) / 256, 256, 0, stream>>>(hodge, sheaf, alpha, beta, out);
}
